// EdgeConvNodeRegressor_11888469475719
// MI455X (gfx1250) — compile-verified, matrix operand order checked by reading, not run on hardware
//
#include <hip/hip_runtime.h>
#include <hip/hip_bf16.h>

// ---------------------------------------------------------------------------
// EdgeConv (DGCNN-style) node regressor for MI455X / gfx1250.
//   per layer: m_e = relu([x_i | x_j - x_i] @ W1 + b1) @ W2 + b2
//              h_i = relu(segment_max_e->i(m_e))   (== max(0, max_e m_e))
//   out_i = h_i @ Wout + bout
// Edge MLP as dense WMMA GEMMs. Weights staged into LDS via the Tensor Data
// Mover (tensor_load_to_lds) once per block, reused across 80 tile-GEMMs.
// ---------------------------------------------------------------------------

#define N_NODES 50000
#define N_EDGES 800000
#define DH      128
#define NLAYERS 3

typedef __attribute__((ext_vector_type(16))) _Float16 v16h;
typedef __attribute__((ext_vector_type(8)))  float    v8f;
typedef unsigned int u32x4 __attribute__((ext_vector_type(4)));
typedef int          i32x8 __attribute__((ext_vector_type(8)));
typedef int          i32x4 __attribute__((ext_vector_type(4)));

// Fragment k order this code uses (contiguous, same in A and B) for V_WMMA_F32_16X16X32_F16:
//   A (16x32 f16): lane<16 -> row M=lane,   K=0..15 ; lane>=16 -> row M=lane-16, K=16..31
//   B (32x16 f16): lane<16 -> col N=lane,   K=0..15 ; lane>=16 -> col N=lane-16, K=16..31
// Pre-converted B fragments stored so lane t reads f16[t*16 .. t*16+15].

#define W1_FRAGS_PER_LAYER (8 * 8)     // ktile(256/32) x ntile(128/16)
#define W2_FRAGS_PER_LAYER (4 * 8)     // ktile(128/32) x ntile
#define FRAG_ELEMS 512                 // 32x16 f16
#define W1F_BYTES (NLAYERS * W1_FRAGS_PER_LAYER * FRAG_ELEMS * 2)
#define W2F_BYTES (NLAYERS * W2_FRAGS_PER_LAYER * FRAG_ELEMS * 2)

// LDS layout (f16 units within dynamic shared):
//   w1s @ 0      : 32768 f16 (64 KB)
//   w2s @ 32768  : 16384 f16 (32 KB)
//   hid @ 49152  : 8 waves x 2 tiles x 2048 f16 (64 KB)
#define LDS_W2_OFF_F16   32768
#define LDS_HID_OFF_F16  49152
#define LDS_TOTAL_BYTES  163840      // 160 KB of the WGP's 320 KB

#define ITERS_PER_WAVE 5             // 5 iterations x 32 edges = 160 edges/wave
// grid: 800000 / (8 waves * 160 edges) = 625 blocks

// ---- native CDNA5 f32 max atomic (no return -> STOREcnt path) --------------
__device__ __forceinline__ void atomic_max_f32(float* p, float v) {
    asm volatile("global_atomic_max_num_f32 %0, %1, off scope:SCOPE_DEV"
                 :: "v"(p), "v"(v) : "memory");
}

// ---- Tensor Data Mover: 1-D dword copy global -> LDS -----------------------
// This toolchain exposes the 6-arg builtin:
//   (u32x4 g0, i32x8 g1, i32x4 g2, i32x4 g3, i32x8 pad, i32 cpol)
#if __has_builtin(__builtin_amdgcn_tensor_load_to_lds)
#define HAVE_TDM 1
__device__ __forceinline__ void tdm_load_1d(const void* gsrc, unsigned lds_byte_off,
                                            unsigned ndwords) {
    unsigned long long ga = (unsigned long long)gsrc;
    u32x4 g0;
    g0.x = 1u;                                               // count=1 user D#
    g0.y = lds_byte_off;                                     // lds_addr
    g0.z = (unsigned)ga;                                     // global_addr[31:0]
    g0.w = (unsigned)((ga >> 32) & 0x1FFFFFFu) | (2u << 30); // ga[56:32] | type=2
    i32x8 g1;
    g1[0] = 0x20000;                                  // data_size=4B, wg_mask=0
    g1[1] = (int)((ndwords & 0xFFFFu) << 16);         // tensor_dim0[15:0]
    g1[2] = (int)(((ndwords >> 16) & 0xFFFFu) | (1u << 16)); // td0 hi | tensor_dim1=1
    g1[3] = (int)(ndwords << 16);                     // tile_dim0
    g1[4] = 1;                                        // tile_dim1=1
    g1[5] = (int)ndwords;                             // tensor_dim0_stride
    g1[6] = 0;
    g1[7] = 0;
    i32x4 z4 = {0, 0, 0, 0};
    i32x8 z8 = {0, 0, 0, 0, 0, 0, 0, 0};
    __builtin_amdgcn_tensor_load_to_lds(g0, g1, z4, z4, z8, 0);
}
#else
#define HAVE_TDM 0
#endif

// ---- weight conversion: f32 row-major -> f16 B-fragment layout -------------
__global__ void convert_w1_kernel(const float* __restrict__ W1, _Float16* __restrict__ w1f) {
    int idx = blockIdx.x * 256 + threadIdx.x;       // 3*64*512 = 98304 threads
    int i    = idx & 15;
    int t    = (idx >> 4) & 31;
    int frag = idx >> 9;                            // l*64 + k*8 + n
    int n = frag & 7, k = (frag >> 3) & 7, l = frag >> 6;
    int K = 32 * k + (t >> 4) * 16 + i;
    int N = 16 * n + (t & 15);
    w1f[idx] = (_Float16)W1[((size_t)l * 256 + K) * DH + N];
}

__global__ void convert_w2_kernel(const float* __restrict__ W2, _Float16* __restrict__ w2f) {
    int idx = blockIdx.x * 256 + threadIdx.x;       // 3*32*512 = 49152 threads
    int i    = idx & 15;
    int t    = (idx >> 4) & 31;
    int frag = idx >> 9;                            // l*32 + k*8 + n
    int n = frag & 7, k = (frag >> 3) & 3, l = frag >> 5;
    int K = 32 * k + (t >> 4) * 16 + i;
    int N = 16 * n + (t & 15);
    w2f[idx] = (_Float16)W2[((size_t)l * DH + K) * DH + N];
}

__global__ void zero_kernel(float4* __restrict__ p, int n4) {
    int i = blockIdx.x * 256 + threadIdx.x;
    if (i < n4) p[i] = make_float4(0.f, 0.f, 0.f, 0.f);
}

// ---- A fragment builder: lane's 16 K-values from the gathered rows ---------
__device__ __forceinline__ v16h build_a_frag(const float* __restrict__ xi,
                                             const float* __restrict__ xj,
                                             int k, int khalf) {
    v16h a;
    const int f = k * 32 + khalf * 16;
    if (k < 4) {                               // first half of feat: x_i
        const float4* p = (const float4*)(xi + f);
        #pragma unroll
        for (int q = 0; q < 4; ++q) {
            float4 v = p[q];
            a[q * 4 + 0] = (_Float16)v.x;  a[q * 4 + 1] = (_Float16)v.y;
            a[q * 4 + 2] = (_Float16)v.z;  a[q * 4 + 3] = (_Float16)v.w;
        }
    } else {                                   // second half: x_j - x_i
        const int f2 = f - DH;
        const float4* pj = (const float4*)(xj + f2);
        const float4* pi = (const float4*)(xi + f2);
        #pragma unroll
        for (int q = 0; q < 4; ++q) {
            float4 vj = pj[q], vi = pi[q];
            a[q * 4 + 0] = (_Float16)(vj.x - vi.x);  a[q * 4 + 1] = (_Float16)(vj.y - vi.y);
            a[q * 4 + 2] = (_Float16)(vj.z - vi.z);  a[q * 4 + 3] = (_Float16)(vj.w - vi.w);
        }
    }
    return a;
}

// ---- main edge kernel: one wave == 2 M-tiles (32 edges) per iteration ------
__global__ __launch_bounds__(256)
void edgeconv_wmma_kernel(const float* __restrict__ hin,
                          const int*   __restrict__ src,
                          const int*   __restrict__ dst,
                          const _Float16* __restrict__ w1f,   // this layer's GEMM1 B frags
                          const float* __restrict__ b1,
                          const _Float16* __restrict__ w2f,   // this layer's GEMM2 B frags
                          const float* __restrict__ b2,
                          float* __restrict__ agg) {
    extern __shared__ __align__(32) _Float16 smem[];
    _Float16* w1s = smem;                       // 64 KB of GEMM1 B fragments
    _Float16* w2s = smem + LDS_W2_OFF_F16;      // 32 KB of GEMM2 B fragments

    const int wave  = threadIdx.x >> 5;
    const int lane  = threadIdx.x & 31;
    const int row   = lane & 15;   // A row / C column index for this lane
    const int khalf = lane >> 4;   // which 16-wide K half this lane holds

    // ---- stage this layer's weights into LDS once per block ---------------
#if HAVE_TDM
    if (wave == 0) {
        tdm_load_1d(w1f, 0u, 16384u);                    // 64 KB
        tdm_load_1d(w2f, LDS_W2_OFF_F16 * 2u, 8192u);    // 32 KB
        __builtin_amdgcn_s_wait_tensorcnt(0);
    }
#else
    {
        const uint4* s1 = (const uint4*)w1f;
        uint4*       d1 = (uint4*)w1s;
        for (int i = threadIdx.x; i < 4096; i += 256) d1[i] = s1[i];
        const uint4* s2 = (const uint4*)w2f;
        uint4*       d2 = (uint4*)w2s;
        for (int i = threadIdx.x; i < 2048; i += 256) d2[i] = s2[i];
    }
#endif
    __syncthreads();

    _Float16* hl0 = smem + LDS_HID_OFF_F16 + wave * 4096;   // per-wave hidden tile 0
    _Float16* hl1 = hl0 + 2048;                             // per-wave hidden tile 1

    const int task   = blockIdx.x * 8 + wave;               // 0..4999
    const int ebase0 = task * (ITERS_PER_WAVE * 32);

    #pragma unroll 1
    for (int it = 0; it < ITERS_PER_WAVE; ++it) {
        const int ebase = ebase0 + it * 32;
        const int e0 = ebase + row;
        const int e1 = ebase + 16 + row;

        const float* xi0 = hin + (size_t)dst[e0] * DH;
        const float* xj0 = hin + (size_t)src[e0] * DH;
        const float* xi1 = hin + (size_t)dst[e1] * DH;
        const float* xj1 = hin + (size_t)src[e1] * DH;

        // -------- GEMM1: [32 x 256] @ [256 x 128], B from LDS --------------
        v8f acc1[2][8] = {};
        #pragma unroll
        for (int k = 0; k < 8; ++k) {
            v16h a0 = build_a_frag(xi0, xj0, k, khalf);
            v16h a1 = build_a_frag(xi1, xj1, k, khalf);
            #pragma unroll
            for (int n = 0; n < 8; ++n) {
                v16h b = *(const v16h*)(w1s + (k * 8 + n) * FRAG_ELEMS + lane * 16);
                acc1[0][n] = __builtin_amdgcn_wmma_f32_16x16x32_f16(
                    false, a0, false, b, (short)0, acc1[0][n], false, false);
                acc1[1][n] = __builtin_amdgcn_wmma_f32_16x16x32_f16(
                    false, a1, false, b, (short)0, acc1[1][n], false, false);
            }
        }

        // bias + ReLU, stash both hidden tiles to per-wave LDS (row-major f16).
        // C/D layout: VGPR r -> (M=r, N=lane) lanes 0-15 ; (M=r+8, N=lane-16).
        #pragma unroll
        for (int n = 0; n < 8; ++n) {
            const int ch = n * 16 + row;
            const float bb = b1[ch];
            #pragma unroll
            for (int r = 0; r < 8; ++r) {
                const int m = r + khalf * 8;
                float v0 = acc1[0][n][r] + bb;
                float v1 = acc1[1][n][r] + bb;
                hl0[m * DH + ch] = (_Float16)(v0 > 0.f ? v0 : 0.f);
                hl1[m * DH + ch] = (_Float16)(v1 > 0.f ? v1 : 0.f);
            }
        }
        // per-wave LDS region: compiler-inserted s_wait_dscnt covers RAW below.

        // -------- GEMM2: [32 x 128] @ [128 x 128], A + B from LDS ----------
        v8f acc2[2][8] = {};
        #pragma unroll
        for (int k = 0; k < 4; ++k) {
            v16h a0 = *(const v16h*)(hl0 + row * DH + k * 32 + khalf * 16);
            v16h a1 = *(const v16h*)(hl1 + row * DH + k * 32 + khalf * 16);
            #pragma unroll
            for (int n = 0; n < 8; ++n) {
                v16h b = *(const v16h*)(w2s + (k * 8 + n) * FRAG_ELEMS + lane * 16);
                acc2[0][n] = __builtin_amdgcn_wmma_f32_16x16x32_f16(
                    false, a0, false, b, (short)0, acc2[0][n], false, false);
                acc2[1][n] = __builtin_amdgcn_wmma_f32_16x16x32_f16(
                    false, a1, false, b, (short)0, acc2[1][n], false, false);
            }
        }

        // -------- bias2 + segment-max scatter (native f32 max atomic) ------
        int nodes0[8], nodes1[8];
        #pragma unroll
        for (int r = 0; r < 8; ++r) {
            nodes0[r] = dst[ebase + r + khalf * 8];
            nodes1[r] = dst[ebase + 16 + r + khalf * 8];
        }
        #pragma unroll
        for (int n = 0; n < 8; ++n) {
            const int ch = n * 16 + row;
            const float bb = b2[ch];
            #pragma unroll
            for (int r = 0; r < 8; ++r) {
                atomic_max_f32(agg + (size_t)nodes0[r] * DH + ch, acc2[0][n][r] + bb);
                atomic_max_f32(agg + (size_t)nodes1[r] * DH + ch, acc2[1][n][r] + bb);
            }
        }
    }
}

// ---- final projection: one wave per node -----------------------------------
__global__ void proj_kernel(const float* __restrict__ h,
                            const float* __restrict__ Wout,
                            const float* __restrict__ bout,
                            float* __restrict__ out) {
    const int node = blockIdx.x * 8 + (threadIdx.x >> 5);
    const int lane = threadIdx.x & 31;
    float s = 0.f;
    #pragma unroll
    for (int q = 0; q < 4; ++q) {
        const int c = lane + q * 32;
        s += h[(size_t)node * DH + c] * Wout[c];
    }
    #pragma unroll
    for (int off = 16; off > 0; off >>= 1) s += __shfl_xor(s, off, 32);
    if (lane == 0) out[node] = s + bout[0];
}

// ---------------------------------------------------------------------------
extern "C" void kernel_launch(void* const* d_in, const int* in_sizes, int n_in,
                              void* d_out, int out_size, void* d_ws, size_t ws_size,
                              hipStream_t stream) {
    const float* x    = (const float*)d_in[0];
    const int*   ei   = (const int*)  d_in[1];
    const float* W1   = (const float*)d_in[2];
    const float* b1   = (const float*)d_in[3];
    const float* W2   = (const float*)d_in[4];
    const float* b2   = (const float*)d_in[5];
    const float* Wout = (const float*)d_in[6];
    const float* bout = (const float*)d_in[7];
    float*       out  = (float*)d_out;

    const int* src = ei;             // edge_index[0]
    const int* dst = ei + N_EDGES;   // edge_index[1]

    char* ws = (char*)d_ws;
    _Float16* w1f = (_Float16*)ws;
    _Float16* w2f = (_Float16*)(ws + W1F_BYTES);
    float*    h0  = (float*)(ws + W1F_BYTES + W2F_BYTES);
    float*    h1  = h0 + (size_t)N_NODES * DH;

    convert_w1_kernel<<<(NLAYERS * W1_FRAGS_PER_LAYER * FRAG_ELEMS) / 256, 256, 0, stream>>>(W1, w1f);
    convert_w2_kernel<<<(NLAYERS * W2_FRAGS_PER_LAYER * FRAG_ELEMS) / 256, 256, 0, stream>>>(W2, w2f);

    float* bufs[2] = {h0, h1};
    const float* hin = x;
    const int n4 = N_NODES * DH / 4;
    const int edge_blocks = N_EDGES / (8 * ITERS_PER_WAVE * 32);   // 625

    for (int l = 0; l < NLAYERS; ++l) {
        float* agg = bufs[l & 1];
        zero_kernel<<<(n4 + 255) / 256, 256, 0, stream>>>((float4*)agg, n4);
        edgeconv_wmma_kernel<<<edge_blocks, 256, LDS_TOTAL_BYTES, stream>>>(
            hin, src, dst,
            w1f + (size_t)l * W1_FRAGS_PER_LAYER * FRAG_ELEMS, b1 + (size_t)l * DH,
            w2f + (size_t)l * W2_FRAGS_PER_LAYER * FRAG_ELEMS, b2 + (size_t)l * DH,
            agg);
        hin = agg;
    }

    proj_kernel<<<N_NODES / 8, 256, 0, stream>>>(hin, Wout, bout, out);
}
